// Cross_AttentionBlock_37151467110634
// MI455X (gfx1250) — matrix operand order repaired and checked by reading, not compiled, not run on hardware
//
#include <hip/hip_runtime.h>

typedef unsigned short u16;
typedef __attribute__((ext_vector_type(16))) __bf16        v16bf;
typedef __attribute__((ext_vector_type(8)))  float         v8f;
typedef __attribute__((ext_vector_type(4)))  unsigned int  v4u;

static constexpr int Bn = 8, Sn = 20, Cn = 512, HWn = 48 * 48;  // 2304
static constexpr int KCn = 256, VCn = 512, OCn = 512;

__device__ __forceinline__ u16 f2bf(float f) {
  unsigned u = __float_as_uint(f);
  unsigned r = u + 0x7fffu + ((u >> 16) & 1u);   // round-to-nearest-even
  return (u16)(r >> 16);
}

union FragBF { v4u q[2]; v16bf v; u16 u[16]; };

// ---------------------------------------------------------------------------
// Tiled bf16 WMMA GEMM:  C[M,N] (+)= alpha * op(A)[M,K] * op(B)[K,N] + bias[M]
// A,B bf16; C f32 or bf16. 256 threads = 8 waves; 128x128 block tile,
// BK=32 k-step; double-buffered LDS; each wave: 32x64 (2x4 wmma frags).
// ---------------------------------------------------------------------------
template<bool TRANSA, bool TRANSB, bool OUT_BF16, bool ACCUM>
__global__ __launch_bounds__(256)
void gemm_wmma_bf16(const u16* __restrict__ A, long sA, int lda,
                    const u16* __restrict__ Bm, long sB, int ldb,
                    void* __restrict__ Cm, long sC,
                    const float* __restrict__ bias,
                    int M, int N, int K, float alpha) {
  constexpr int BM = 128, BN = 128, BK = 32, LDK = 40;  // 40: +8 pad, keeps 16B align
  __shared__ alignas(16) u16 At[2][BM][LDK];
  __shared__ alignas(16) u16 Bt[2][BN][LDK];            // stored [n][k]

  const int bz = blockIdx.z;
  const int m0 = blockIdx.y * BM;
  const int n0 = blockIdx.x * BN;
  A  += (long)bz * sA;
  Bm += (long)bz * sB;

  const int tid  = threadIdx.x;
  const int lane = tid & 31;
  const int wave = tid >> 5;
  const int wm   = (wave & 3) * 32;   // wave row offset inside tile
  const int wn   = (wave >> 2) * 64;  // wave col offset inside tile

  v8f acc[2][4];
#pragma unroll
  for (int mi = 0; mi < 2; ++mi)
#pragma unroll
    for (int ni = 0; ni < 4; ++ni)
#pragma unroll
      for (int r = 0; r < 8; ++r) acc[mi][ni][r] = 0.0f;

  // Per-thread global-load coordinates (16 bf16 = 2 x b128 each for A and B)
  int a_r, a_c;
  if (!TRANSA) { a_r = tid >> 1;        a_c = (tid & 1) * 16; }  // row, k-start
  else         { a_c = tid >> 3;        a_r = (tid & 7) * 16; }  // k,   m-chunk
  int b_r, b_c;
  if (!TRANSB) { b_c = tid >> 3;        b_r = (tid & 7) * 16; }  // k,   n-chunk
  else         { b_r = tid >> 1;        b_c = (tid & 1) * 16; }  // n,   k-start

  auto fetchA = [&](int k0, v4u& x0, v4u& x1) {
    const u16* p = !TRANSA ? A + (long)(m0 + a_r) * lda + (k0 + a_c)
                           : A + (long)(k0 + a_c) * lda + (m0 + a_r);
    const v4u* q4 = reinterpret_cast<const v4u*>(p);
    x0 = q4[0]; x1 = q4[1];
  };
  auto fetchB = [&](int k0, v4u& x0, v4u& x1) {
    const u16* p = !TRANSB ? Bm + (long)(k0 + b_c) * ldb + (n0 + b_r)
                           : Bm + (long)(n0 + b_r) * ldb + (k0 + b_c);
    const v4u* q4 = reinterpret_cast<const v4u*>(p);
    x0 = q4[0]; x1 = q4[1];
  };
  auto stashA = [&](int buf, v4u x0, v4u x1) {
    if (!TRANSA) {
      *reinterpret_cast<v4u*>(&At[buf][a_r][a_c])     = x0;
      *reinterpret_cast<v4u*>(&At[buf][a_r][a_c + 8]) = x1;
    } else {
      FragBF t; t.q[0] = x0; t.q[1] = x1;
#pragma unroll
      for (int j = 0; j < 16; ++j) At[buf][a_r + j][a_c] = t.u[j];
    }
  };
  auto stashB = [&](int buf, v4u x0, v4u x1) {
    if (!TRANSB) {
      FragBF t; t.q[0] = x0; t.q[1] = x1;
#pragma unroll
      for (int j = 0; j < 16; ++j) Bt[buf][b_r + j][b_c] = t.u[j];
    } else {
      *reinterpret_cast<v4u*>(&Bt[buf][b_r][b_c])     = x0;
      *reinterpret_cast<v4u*>(&Bt[buf][b_r][b_c + 8]) = x1;
    }
  };

  { v4u a0, a1, b0, b1;
    fetchA(0, a0, a1); fetchB(0, b0, b1);
    stashA(0, a0, a1); stashB(0, b0, b1); }
  __syncthreads();

  const int nk   = K / BK;
  const int arow = wm + (lane & 15);
  const int klo  = (lane & 16) ? 8 : 0;    // A K-half per lane group
  const int kbs  = (lane & 16) ? 8 : 0;    // B K-base per lane group
  const int bcol = wn + (lane & 15);

  for (int kt = 0; kt < nk; ++kt) {
    const int cur = kt & 1;
    v4u a0, a1, b0, b1;
    const bool more = (kt + 1) < nk;
    if (more) { fetchA((kt + 1) * BK, a0, a1); fetchB((kt + 1) * BK, b0, b1); }

    FragBF af[2], bfr[4];
#pragma unroll
    for (int mi = 0; mi < 2; ++mi) {
      const u16* s = &At[cur][arow + mi * 16][0];
      af[mi].q[0] = *reinterpret_cast<const v4u*>(s + klo);
      af[mi].q[1] = *reinterpret_cast<const v4u*>(s + 16 + klo);
    }
#pragma unroll
    for (int ni = 0; ni < 4; ++ni) {
      const u16* s = &Bt[cur][bcol + ni * 16][0];
      bfr[ni].q[0] = *reinterpret_cast<const v4u*>(s + kbs);
      bfr[ni].q[1] = *reinterpret_cast<const v4u*>(s + 16 + kbs);
    }
#pragma unroll
    for (int mi = 0; mi < 2; ++mi)
#pragma unroll
      for (int ni = 0; ni < 4; ++ni)
        acc[mi][ni] = __builtin_amdgcn_wmma_f32_16x16x32_bf16(
            false, af[mi].v, false, bfr[ni].v, (short)0, acc[mi][ni], false, false);

    if (more) { stashA(cur ^ 1, a0, a1); stashB(cur ^ 1, b0, b1); }
    __syncthreads();
  }

  // Epilogue: D layout — col = lane&15, rows r / 8+r per lane half.
  const int colb = n0 + wn + (lane & 15);
  const int rowb = m0 + wm + ((lane & 16) ? 8 : 0);
#pragma unroll
  for (int mi = 0; mi < 2; ++mi) {
#pragma unroll
    for (int ni = 0; ni < 4; ++ni) {
      const int col = colb + ni * 16;
#pragma unroll
      for (int r = 0; r < 8; ++r) {
        const int row = rowb + mi * 16 + r;
        float v = alpha * acc[mi][ni][r];
        if (bias) v += bias[row];
        const long idx = (long)row * N + col;
        if (OUT_BF16) {
          reinterpret_cast<u16*>(Cm)[(long)bz * sC + idx] = f2bf(v);
        } else {
          float* O = reinterpret_cast<float*>(Cm) + (long)bz * sC;
          if (ACCUM) v += O[idx];
          O[idx] = v;
        }
      }
    }
  }
}

// ---------------------------------------------------------------------------
// Small elementwise / word-attention kernels
// ---------------------------------------------------------------------------
__global__ void k_cvt_bf16(const float* __restrict__ in, u16* __restrict__ out, long n) {
  long i = (long)blockIdx.x * blockDim.x + threadIdx.x;
  if (i < n) out[i] = f2bf(in[i]);
}

__global__ void k_word_logits(const float* __restrict__ query,
                              const float* __restrict__ vis,
                              float* __restrict__ attn) {
  const int b = blockIdx.y;
  const int n = blockIdx.x * blockDim.x + threadIdx.x;
  const float* vb = vis   + (long)b * Cn * HWn;
  const float* qb = query + (long)b * Sn * Cn;
  float acc[Sn];
#pragma unroll
  for (int s = 0; s < Sn; ++s) acc[s] = 0.f;
  for (int c = 0; c < Cn; ++c) {
    const float vv = vb[(long)c * HWn + n];
#pragma unroll
    for (int s = 0; s < Sn; ++s) acc[s] += qb[s * Cn + c] * vv;
  }
#pragma unroll
  for (int s = 0; s < Sn; ++s) attn[((long)b * Sn + s) * HWn + n] = acc[s];
}

__global__ void k_word_softmax(float* __restrict__ attn, const int* __restrict__ wid) {
  const int b = blockIdx.y;
  const int n = blockIdx.x * blockDim.x + threadIdx.x;
  float v[Sn];
  float mx = -3.4e38f;
#pragma unroll
  for (int s = 0; s < Sn; ++s) { v[s] = attn[((long)b * Sn + s) * HWn + n]; mx = fmaxf(mx, v[s]); }
  float sum = 0.f;
#pragma unroll
  for (int s = 0; s < Sn; ++s) {
    float e = __expf(v[s] - mx);
    e *= (wid[b * Sn + s] != 0) ? 1.f : 0.f;   // mask + renorm == masked softmax
    v[s] = e; sum += e;
  }
  const float inv = 1.f / sum;
#pragma unroll
  for (int s = 0; s < Sn; ++s) attn[((long)b * Sn + s) * HWn + n] = v[s] * inv;
}

__global__ void k_wemb(const float* __restrict__ query,
                       const float* __restrict__ attn,
                       u16* __restrict__ wembb) {
  const int b = blockIdx.z, c = blockIdx.y;
  const int n = blockIdx.x * blockDim.x + threadIdx.x;
  float acc = 0.f;
#pragma unroll
  for (int s = 0; s < Sn; ++s)
    acc += query[((long)b * Sn + s) * Cn + c] * attn[((long)b * Sn + s) * HWn + n];
  wembb[((long)b * Cn + c) * HWn + n] = f2bf(acc);
}

__global__ void k_row_softmax(const float* __restrict__ sim, u16* __restrict__ simb) {
  const long row = blockIdx.x;
  const float* r = sim + row * HWn;
  __shared__ float red[256];
  const int t = threadIdx.x;
  float mx = -3.4e38f;
  for (int j = t; j < HWn; j += 256) mx = fmaxf(mx, r[j]);
  red[t] = mx; __syncthreads();
  for (int o = 128; o > 0; o >>= 1) { if (t < o) red[t] = fmaxf(red[t], red[t + o]); __syncthreads(); }
  mx = red[0]; __syncthreads();
  float sum = 0.f;
  for (int j = t; j < HWn; j += 256) sum += __expf(r[j] - mx);
  red[t] = sum; __syncthreads();
  for (int o = 128; o > 0; o >>= 1) { if (t < o) red[t] += red[t + o]; __syncthreads(); }
  const float inv = 1.f / red[0];
  for (int j = t; j < HWn; j += 256) simb[row * HWn + j] = f2bf(__expf(r[j] - mx) * inv);
}

__global__ void k_bias_sum(const float* __restrict__ a, const float* __restrict__ b,
                           float* __restrict__ o, int n) {
  int i = blockIdx.x * blockDim.x + threadIdx.x;
  if (i < n) o[i] = a[i] + b[i];
}

// ---------------------------------------------------------------------------
extern "C" void kernel_launch(void* const* d_in, const int* in_sizes, int n_in,
                              void* d_out, int out_size, void* d_ws, size_t ws_size,
                              hipStream_t stream) {
  const float* key   = (const float*)d_in[0];
  const float* query = (const float*)d_in[1];
  const int*   wid   = (const int*)  d_in[2];
  const float* wv_w  = (const float*)d_in[3];
  const float* wv_b  = (const float*)d_in[4];
  const float* fk_w  = (const float*)d_in[5];
  const float* fk_b  = (const float*)d_in[6];
  const float* fq_w  = (const float*)d_in[7];
  const float* fq_b  = (const float*)d_in[8];
  const float* fv_w  = (const float*)d_in[9];
  const float* fv_b  = (const float*)d_in[10];
  const float* w1_w  = (const float*)d_in[11];
  const float* w1_b  = (const float*)d_in[12];
  const float* w2_w  = (const float*)d_in[13];
  const float* w2_b  = (const float*)d_in[14];
  float* out = (float*)d_out;

  char* ws = (char*)d_ws; size_t off = 0;
  auto alloc = [&](size_t bytes) -> char* {
    off = (off + 255) & ~(size_t)255;
    char* p = ws + off; off += bytes; return p;
  };
  u16*   keyb  = (u16*)  alloc((size_t)Bn * Cn * HWn * 2);
  u16*   wvb   = (u16*)  alloc((size_t)Cn * Cn * 2);
  u16*   fkb   = (u16*)  alloc((size_t)KCn * Cn * 2);
  u16*   fqb   = (u16*)  alloc((size_t)KCn * Cn * 2);
  u16*   fvb   = (u16*)  alloc((size_t)VCn * Cn * 2);
  u16*   w1b   = (u16*)  alloc((size_t)OCn * 2 * Cn * 2);
  u16*   w2b   = (u16*)  alloc((size_t)OCn * Cn * 2);
  float* vis   = (float*)alloc((size_t)Bn * Cn * HWn * 4);
  float* attn  = (float*)alloc((size_t)Bn * Sn * HWn * 4);
  u16*   wembb = (u16*)  alloc((size_t)Bn * Cn * HWn * 2);
  u16*   qb    = (u16*)  alloc((size_t)Bn * KCn * HWn * 2);
  u16*   kb    = (u16*)  alloc((size_t)Bn * KCn * HWn * 2);
  u16*   vb    = (u16*)  alloc((size_t)Bn * VCn * HWn * 2);
  float* sim   = (float*)alloc((size_t)Bn * HWn * HWn * 4);
  u16*   simb  = (u16*)  alloc((size_t)Bn * HWn * HWn * 2);
  u16*   ctxb  = (u16*)  alloc((size_t)Bn * VCn * HWn * 2);
  float* bsum  = (float*)alloc((size_t)OCn * 4);

  auto cvt = [&](const float* s, u16* d, long n) {
    k_cvt_bf16<<<dim3((unsigned)((n + 255) / 256)), 256, 0, stream>>>(s, d, n);
  };
  cvt(key,  keyb, (long)Bn * Cn * HWn);
  cvt(wv_w, wvb,  (long)Cn * Cn);
  cvt(fk_w, fkb,  (long)KCn * Cn);
  cvt(fq_w, fqb,  (long)KCn * Cn);
  cvt(fv_w, fvb,  (long)VCn * Cn);
  cvt(w1_w, w1b,  (long)OCn * 2 * Cn);
  cvt(w2_w, w2b,  (long)OCn * Cn);

  const dim3 blk(256);
  // vis = wv * key + wv_b                                  (f32)
  gemm_wmma_bf16<false, false, false, false><<<dim3(HWn / 128, Cn / 128, Bn), blk, 0, stream>>>(
      wvb, 0, Cn, keyb, (long)Cn * HWn, HWn, vis, (long)Cn * HWn, wv_b, Cn, HWn, Cn, 1.0f);

  k_word_logits<<<dim3(HWn / 256, Bn), blk, 0, stream>>>(query, vis, attn);
  k_word_softmax<<<dim3(HWn / 256, Bn), blk, 0, stream>>>(attn, wid);
  k_wemb<<<dim3(HWn / 256, Cn, Bn), blk, 0, stream>>>(query, attn, wembb);

  // k = fk*key, q = fq*wemb, v = fv*key                    (bf16)
  gemm_wmma_bf16<false, false, true, false><<<dim3(HWn / 128, KCn / 128, Bn), blk, 0, stream>>>(
      fkb, 0, Cn, keyb, (long)Cn * HWn, HWn, kb, (long)KCn * HWn, fk_b, KCn, HWn, Cn, 1.0f);
  gemm_wmma_bf16<false, false, true, false><<<dim3(HWn / 128, KCn / 128, Bn), blk, 0, stream>>>(
      fqb, 0, Cn, wembb, (long)Cn * HWn, HWn, qb, (long)KCn * HWn, fq_b, KCn, HWn, Cn, 1.0f);
  gemm_wmma_bf16<false, false, true, false><<<dim3(HWn / 128, VCn / 128, Bn), blk, 0, stream>>>(
      fvb, 0, Cn, keyb, (long)Cn * HWn, HWn, vb, (long)VCn * HWn, fv_b, VCn, HWn, Cn, 1.0f);

  // sim = (q^T k) * KC^-0.5                                (f32)
  gemm_wmma_bf16<true, false, false, false><<<dim3(HWn / 128, HWn / 128, Bn), blk, 0, stream>>>(
      qb, (long)KCn * HWn, HWn, kb, (long)KCn * HWn, HWn, sim, (long)HWn * HWn, nullptr,
      HWn, HWn, KCn, 0.0625f);

  k_row_softmax<<<dim3(Bn * HWn), blk, 0, stream>>>(sim, simb);

  // ctx = v * softmax(sim)^T                               (bf16)
  gemm_wmma_bf16<false, true, true, false><<<dim3(HWn / 128, VCn / 128, Bn), blk, 0, stream>>>(
      vb, (long)VCn * HWn, HWn, simb, (long)HWn * HWn, HWn, ctxb, (long)VCn * HWn, nullptr,
      VCn, HWn, HWn, 1.0f);

  k_bias_sum<<<dim3(2), blk, 0, stream>>>(w1_b, w2_b, bsum, OCn);

  // out = w1[:, :C]*ctx + (b1+b2); += w1[:, C:]*wemb; += w2*ctx
  gemm_wmma_bf16<false, false, false, false><<<dim3(HWn / 128, OCn / 128, Bn), blk, 0, stream>>>(
      w1b, 0, 2 * Cn, ctxb, (long)VCn * HWn, HWn, out, (long)OCn * HWn, bsum, OCn, HWn, VCn, 1.0f);
  gemm_wmma_bf16<false, false, false, true><<<dim3(HWn / 128, OCn / 128, Bn), blk, 0, stream>>>(
      w1b + Cn, 0, 2 * Cn, wembb, (long)Cn * HWn, HWn, out, (long)OCn * HWn, nullptr, OCn, HWn, Cn, 1.0f);
  gemm_wmma_bf16<false, false, false, true><<<dim3(HWn / 128, OCn / 128, Bn), blk, 0, stream>>>(
      w2b, 0, Cn, ctxb, (long)VCn * HWn, HWn, out, (long)OCn * HWn, nullptr, OCn, HWn, VCn, 1.0f);

  (void)in_sizes; (void)n_in; (void)out_size; (void)ws_size;
}
